// LinearSpatial_55353538511199
// MI455X (gfx1250) — matrix operand order repaired and checked by reading, not compiled, not run on hardware
//
#include <hip/hip_runtime.h>

typedef __attribute__((ext_vector_type(16))) __bf16 v16bf;
typedef __attribute__((ext_vector_type(8)))  __bf16 v8bf;
typedef __attribute__((ext_vector_type(8)))  float  v8f;

#define F 16
#define OUTC 512
#define KP 288              // 272 padded to 9*32
#define ASTRIDE 296         // LDS row stride (elems): 148 dwords -> conflict-free b128 pattern

// ---------------- zero agg ----------------
__global__ void zero_f32(float* __restrict__ p, long n) {
    long i = (long)blockIdx.x * blockDim.x + threadIdx.x;
    if (i < n) p[i] = 0.0f;
}

// ---------------- edge scatter: agg[src] += x[dst] ----------------
// one thread per (edge, feature): coalesced x reads, scalar f32 atomics (L2-resident target)
__global__ void edge_scatter(const long long* __restrict__ ei,
                             const float* __restrict__ x,
                             float* __restrict__ agg, long E) {
    long gid = (long)blockIdx.x * blockDim.x + threadIdx.x;
    if (gid >= E * F) return;
    long e = gid >> 4;
    int  f = (int)(gid & 15);
    long s = ei[e];        // src
    long d = ei[E + e];    // dst
    atomicAdd(&agg[s * F + f], x[d * F + f]);
}

// ---------------- pack W (272x512 f32, row-major) -> Wcm (512xKP bf16, col-major K) ----------------
__global__ void pack_w(const float* __restrict__ W, __bf16* __restrict__ Wcm) {
    int col = blockIdx.x;          // 0..511
    int k   = threadIdx.x;         // 0..KP-1
    Wcm[(size_t)col * KP + k] = (k < 272) ? (__bf16)W[(size_t)k * OUTC + col] : (__bf16)0.0f;
}

// ---------------- fused expand + GEMM ----------------
// one block (8 waves) per 16-row block; builds A-tile (16 x KP bf16) in LDS, then
// wave w computes cols [64w, 64w+64) with v_wmma_f32_16x16x32_bf16 (9 K-steps x 4 N-tiles).
__global__ void __launch_bounds__(256)
expand_gemm(const float* __restrict__ x,
            const float* __restrict__ agg,
            const __bf16* __restrict__ Wcm,
            const float* __restrict__ bias,
            float* __restrict__ out) {
    __shared__ __bf16  xsh[16][F];
    __shared__ unsigned msh[16];
    __shared__ __bf16  Ash[16][ASTRIDE];

    const int  tid      = threadIdx.x;
    const long base_row = (long)blockIdx.x * 16;

    // phase 0: load x row tile, compute sign mask of (x+agg) via wave32 ballot
    {
        int r = tid >> 4;          // lanes 0-15 -> even row of pair, 16-31 -> odd row
        int c = tid & 15;
        long row = base_row + r;
        float xv = x[row * F + c];
        float xs = xv + agg[row * F + c];
        xsh[r][c] = (__bf16)xv;
        unsigned bal = (unsigned)__ballot(xs > 0.0f);
        if ((tid & 31) == 0) {
            msh[r]     = bal & 0xFFFFu;
            msh[r + 1] = (bal >> 16) & 0xFFFFu;
        }
    }
    __syncthreads();

    // phase 1: build A[r][k]: k<16 -> x ; 16..271 -> x[c1]*mask[c2] ; pad -> 0
    {
        int r = tid & 15, c1 = tid >> 4;
        __bf16 xv = xsh[r][c1];
        unsigned m = msh[r];
        Ash[r][c1] = xv;                       // k = c1
        int kb = 16 + c1 * 16;
        #pragma unroll
        for (int c2 = 0; c2 < 16; ++c2)
            Ash[r][kb + c2] = ((m >> c2) & 1u) ? xv : (__bf16)0.0f;
        Ash[r][272 + c1] = (__bf16)0.0f;       // pad 272..287
        if (c1 < 8) Ash[r][288 + c1] = (__bf16)0.0f; // pad to ASTRIDE region used by loads
    }
    __syncthreads();

    // phase 2: WMMA GEMM
    const int lane = tid & 31;
    const int w    = tid >> 5;       // wave id: cols [64w, 64w+64)
    const int mrow = lane & 15;      // A row for this lane
    const int lh   = lane >> 4;      // lane-half
    const int nloc = lane & 15;      // B/D column-in-tile

    v8f acc[4];
    #pragma unroll
    for (int t = 0; t < 4; ++t) acc[t] = (v8f){0,0,0,0,0,0,0,0};

    #pragma unroll
    for (int ks = 0; ks < 9; ++ks) {
        // A operand: lane half lh -> K = ks*32 + 16*lh + {0..15}
        int o0 = ks * 32 + lh * 16;
        v8bf a0 = *(const v8bf*)&Ash[mrow][o0];
        v8bf a1 = *(const v8bf*)&Ash[mrow][o0 + 8];
        v16bf av;
        #pragma unroll
        for (int i = 0; i < 8; ++i) { av[i] = a0[i]; av[8 + i] = a1[i]; }

        int kb = ks * 32 + lh * 16;  // B: 16 contiguous K per lane half
        #pragma unroll
        for (int t = 0; t < 4; ++t) {
            int col = w * 64 + t * 16 + nloc;
            v16bf bv = *(const v16bf*)(Wcm + (size_t)col * KP + kb);
            acc[t] = __builtin_amdgcn_wmma_f32_16x16x32_bf16(
                false, av, false, bv, (short)0, acc[t], false, false);
        }
    }

    // epilogue: D layout -> row = base + r + 8*lh, col = 64w + 16t + nloc
    #pragma unroll
    for (int t = 0; t < 4; ++t) {
        int col = w * 64 + t * 16 + nloc;
        float bv = bias[col];
        #pragma unroll
        for (int r = 0; r < 8; ++r) {
            long row = base_row + r + 8 * lh;
            out[row * (long)OUTC + col] = acc[t][r] + bv;
        }
    }
}

extern "C" void kernel_launch(void* const* d_in, const int* in_sizes, int n_in,
                              void* d_out, int out_size, void* d_ws, size_t ws_size,
                              hipStream_t stream) {
    const float*     x  = (const float*)d_in[0];
    const long long* ei = (const long long*)d_in[1];
    const float*     W  = (const float*)d_in[2];
    const float*     b  = (const float*)d_in[3];
    float*           out = (float*)d_out;

    const long N = in_sizes[0] / F;       // 200000
    const long E = in_sizes[1] / 2;       // 6400000

    // workspace layout
    float*  agg = (float*)d_ws;                                   // N*16 f32
    size_t  off = (size_t)N * F * sizeof(float);
    off = (off + 63) & ~(size_t)63;
    __bf16* Wcm = (__bf16*)((char*)d_ws + off);                   // 512*KP bf16

    // 1) zero agg
    {
        long n = N * F;
        zero_f32<<<(unsigned)((n + 255) / 256), 256, 0, stream>>>(agg, n);
    }
    // 2) edge scatter
    {
        long n = E * F;
        edge_scatter<<<(unsigned)((n + 255) / 256), 256, 0, stream>>>(ei, x, agg, E);
    }
    // 3) pack W
    pack_w<<<OUTC, KP, 0, stream>>>(W, Wcm);
    // 4) fused expand + WMMA GEMM
    expand_gemm<<<(unsigned)(N / 16), 256, 0, stream>>>(x, agg, Wcm, b, out);
}
